// GNN_8237747274113
// MI455X (gfx1250) — matrix operand order repaired and checked by reading, not compiled, not run on hardware
//
#include <hip/hip_runtime.h>
#include <cstdint>

typedef __attribute__((ext_vector_type(16))) __bf16 bf16x16;
typedef __attribute__((ext_vector_type(8)))  __bf16 bf16x8;
typedef __attribute__((ext_vector_type(8)))  float  f32x8;

__device__ __forceinline__ uint16_t f2bf(float f) {
  uint32_t u = __float_as_uint(f);
  uint32_t r = u + 0x7FFFu + ((u >> 16) & 1u);   // round-to-nearest-even
  return (uint16_t)(r >> 16);
}

// ---------------------------------------------------------------------------
// Generic fill
// ---------------------------------------------------------------------------
__global__ void fill_u32(unsigned* __restrict__ p, unsigned val, long long n) {
  long long i = (long long)blockIdx.x * blockDim.x + threadIdx.x;
  if (i < n) p[i] = val;
}

// ---------------------------------------------------------------------------
// Weight convert: W [K x Ncols] f32 row-major -> Wt [Ncols x Kpad] bf16 (zero pad)
// ---------------------------------------------------------------------------
__global__ void convert_weight(const float* __restrict__ W, uint16_t* __restrict__ Wt,
                               int K, int Ncols, int Kpad) {
  int idx = blockIdx.x * blockDim.x + threadIdx.x;
  if (idx >= Ncols * Kpad) return;
  int n = idx / Kpad, k = idx % Kpad;
  Wt[idx] = (k < K) ? f2bf(W[(size_t)k * Ncols + n]) : (uint16_t)0;
}

// ---------------------------------------------------------------------------
// h0 = [x | renorm(gene_table[pos])] -> bf16 rows, Kpad=192.  One wave / node.
// ---------------------------------------------------------------------------
__global__ void build_h0(const float* __restrict__ x, const int* __restrict__ pos,
                         const float* __restrict__ gene_table,
                         uint16_t* __restrict__ SB, int N) {
  int wave = (blockIdx.x * blockDim.x + threadIdx.x) >> 5;
  int lane = threadIdx.x & 31;
  if (wave >= N) return;
  const float* gt = gene_table + (size_t)pos[wave] * 128;
  float ss = 0.f;
#pragma unroll
  for (int j = 0; j < 4; ++j) { float g = gt[lane + 32 * j]; ss += g * g; }
#pragma unroll
  for (int off = 16; off; off >>= 1) ss += __shfl_xor(ss, off, 32);
  float nrm = sqrtf(ss);
  float scale = (nrm > 1.0f) ? 1.0f / (nrm + 1e-7f) : 1.0f;
  uint16_t* row = SB + (size_t)wave * 192;
  const float* xr = x + (size_t)wave * 64;
#pragma unroll
  for (int j = 0; j < 6; ++j) {
    int c = lane + 32 * j;
    float v = (c < 64) ? xr[c] : gt[c - 64] * scale;
    row[c] = f2bf(v);
  }
}

// ---------------------------------------------------------------------------
// WMMA bf16 GEMM, LDS-staged weights:
//  - whole Bt [NT*16 x Kpad] is copied to LDS once per block (+8-half row pad
//    to destagger the 64 LDS banks); all 8 waves share it.
//  - one wave computes a full 16 x (NT*16) C stripe: A fragment loaded once
//    per K-step (global), all NT B fragments fetched from LDS into registers,
//    then NT back-to-back v_wmma_f32_16x16x32_bf16.
// ---------------------------------------------------------------------------
template <int NT>
__global__ void wmma_gemm_lds(const uint16_t* __restrict__ A,
                              const uint16_t* __restrict__ Bt,
                              const float* __restrict__ bias,
                              float* __restrict__ C,
                              int M, int Kpad) {
  extern __shared__ __align__(16) uint16_t sB[];
  const int sKpad = Kpad + 8;                       // bank-destagger pad (16 B)

  // cooperative stage of Bt into LDS (16 B per thread per chunk)
  const int chunksPerRow = Kpad >> 3;               // Kpad % 8 == 0
  const int nchunks = NT * 16 * chunksPerRow;
  for (int c = threadIdx.x; c < nchunks; c += blockDim.x) {
    int row = c / chunksPerRow, off = (c % chunksPerRow) << 3;
    *(uint4*)(sB + (size_t)row * sKpad + off) =
        *(const uint4*)(Bt + (size_t)row * Kpad + off);
  }
  __syncthreads();

  int wave = (int)(((long long)blockIdx.x * blockDim.x + threadIdx.x) >> 5);
  int lane = threadIdx.x & 31;
  int mtile = (M + 15) >> 4;
  if (wave >= mtile) return;
  int tm = wave << 4;
  int l  = lane & 15, hs = lane >> 4;

  int ra = tm + l; if (ra >= M) ra = M - 1;         // clamp for ragged last tile
  const uint16_t* arow = A + (size_t)ra * Kpad;

  f32x8 acc[NT];
#pragma unroll
  for (int t = 0; t < NT; ++t) acc[t] = (f32x8){};

  for (int k0 = 0; k0 < Kpad; k0 += 32) {
    // A fragment: lane(M=l) holds K = {k0+hs*8 .. +7} and {k0+16+hs*8 .. +7}
    bf16x8 a0 = *(const bf16x8*)(arow + k0 + hs * 8);
    bf16x8 a1 = *(const bf16x8*)(arow + k0 + 16 + hs * 8);
    bf16x16 av = __builtin_shufflevector(a0, a1, 0,1,2,3,4,5,6,7,8,9,10,11,12,13,14,15);

    // fetch ALL B fragments from LDS first ...
    bf16x16 bv[NT];
#pragma unroll
    for (int t = 0; t < NT; ++t) {
      const uint16_t* brow = sB + (size_t)(t * 16 + l) * sKpad + k0 + hs * 8;
      bf16x8 b0 = *(const bf16x8*)(brow);
      bf16x8 b1 = *(const bf16x8*)(brow + 16);
      bv[t] = __builtin_shufflevector(b0, b1, 0,1,2,3,4,5,6,7,8,9,10,11,12,13,14,15);
    }
    // ... then issue NT WMMAs back-to-back (single wait ahead of the clause)
#pragma unroll
    for (int t = 0; t < NT; ++t)
      acc[t] = __builtin_amdgcn_wmma_f32_16x16x32_bf16(false, av, false, bv[t],
                                                       (short)0, acc[t], false, false);
  }

  const int Ncols = NT * 16;
#pragma unroll
  for (int t = 0; t < NT; ++t) {
    float bvv = bias ? bias[t * 16 + l] : 0.0f;
#pragma unroll
    for (int r = 0; r < 8; ++r) {
      int rr = tm + 8 * hs + r;                     // C layout: VGPR r -> row r + 8*hs
      if (rr < M) C[(size_t)rr * Ncols + t * 16 + l] = acc[t][r] + bvv;
    }
  }
}

// ---------------------------------------------------------------------------
// Per-node attention dots: asrc = H.a_src ; adst = H.a_dst.  One wave / node.
// ---------------------------------------------------------------------------
__global__ void row_dots(const float* __restrict__ H, const float* __restrict__ as,
                         const float* __restrict__ ad,
                         float* __restrict__ asrc, float* __restrict__ adst, int N) {
  int wave = (blockIdx.x * blockDim.x + threadIdx.x) >> 5;
  int lane = threadIdx.x & 31;
  if (wave >= N) return;
  const float* row = H + (size_t)wave * 128;
  float s0 = 0.f, s1 = 0.f;
#pragma unroll
  for (int j = 0; j < 4; ++j) {
    float h = row[lane + 32 * j];
    s0 += h * as[lane + 32 * j];
    s1 += h * ad[lane + 32 * j];
  }
#pragma unroll
  for (int off = 16; off; off >>= 1) { s0 += __shfl_down(s0, off, 32); s1 += __shfl_down(s1, off, 32); }
  if (lane == 0) { asrc[wave] = s0; adst[wave] = s1; }
}

// ---------------------------------------------------------------------------
// Edge softmax passes (edge ids >= E are self-loops)
// ---------------------------------------------------------------------------
__device__ __forceinline__ float leaky(float v) { return v > 0.f ? v : 0.2f * v; }
__device__ __forceinline__ unsigned enc(float f) {
  unsigned u = __float_as_uint(f);
  return u ^ ((u >> 31) ? 0xFFFFFFFFu : 0x80000000u);
}
__device__ __forceinline__ float dec(unsigned k) {
  return __uint_as_float(k ^ ((k >> 31) ? 0x80000000u : 0xFFFFFFFFu));
}

__global__ void edge_max(const int* __restrict__ srcI, const int* __restrict__ dstI,
                         const float* __restrict__ asrc, const float* __restrict__ adst,
                         unsigned* __restrict__ mkey, int E, int N) {
  int e = blockIdx.x * blockDim.x + threadIdx.x;
  if (e >= E + N) return;
  int s = (e < E) ? srcI[e] : (e - E);
  int d = (e < E) ? dstI[e] : (e - E);
  atomicMax(&mkey[d], enc(leaky(asrc[s] + adst[d])));
}

__global__ void edge_sum(const int* __restrict__ srcI, const int* __restrict__ dstI,
                         const float* __restrict__ asrc, const float* __restrict__ adst,
                         const unsigned* __restrict__ mkey, float* __restrict__ denom,
                         int E, int N) {
  int e = blockIdx.x * blockDim.x + threadIdx.x;
  if (e >= E + N) return;
  int s = (e < E) ? srcI[e] : (e - E);
  int d = (e < E) ? dstI[e] : (e - E);
  float v = leaky(asrc[s] + adst[d]);
  atomicAdd(&denom[d], __expf(v - dec(mkey[d])));
}

// Grid-stride wave loop: scatter h[src]*alpha into Gout[dst] (4 atomics/lane),
// prefetching the NEXT edge's source row (global_prefetch_b8) to hide the
// dependent index->row gather latency in this HBM/L2-bound loop.
__global__ void edge_scatter(const int* __restrict__ srcI, const int* __restrict__ dstI,
                             const float* __restrict__ asrc, const float* __restrict__ adst,
                             const unsigned* __restrict__ mkey, const float* __restrict__ denom,
                             const float* __restrict__ H, float* __restrict__ Gout,
                             int E, int N) {
  int lane  = threadIdx.x & 31;
  long long wid = ((long long)blockIdx.x * blockDim.x + threadIdx.x) >> 5;
  long long nw  = (long long)gridDim.x * (blockDim.x >> 5);
  long long tot = (long long)E + N;
  for (long long e = wid; e < tot; e += nw) {
    long long en = e + nw;
    if (en < tot) {                       // prefetch next source row (16 B/lane -> full 512 B row)
      int sn = (en < (long long)E) ? srcI[en] : (int)(en - E);
      __builtin_prefetch(H + (size_t)sn * 128 + lane * 4, 0, 0);
    }
    int s = (e < (long long)E) ? srcI[e] : (int)(e - E);
    int d = (e < (long long)E) ? dstI[e] : (int)(e - E);
    float v = leaky(asrc[s] + adst[d]);
    float alpha = __expf(v - dec(mkey[d])) / (denom[d] + 1e-16f);
    const float* hs = H + (size_t)s * 128;
    float* gd = Gout + (size_t)d * 128;
#pragma unroll
    for (int j = 0; j < 4; ++j)
      atomicAdd(&gd[lane + 32 * j], hs[lane + 32 * j] * alpha);
  }
}

// ---------------------------------------------------------------------------
// h_next = (relu?)(Gout + b + Lb); optionally also emit bf16 copy (Kpad=128)
// ---------------------------------------------------------------------------
__global__ void combine(const float* __restrict__ Gout, const float* __restrict__ Lb,
                        const float* __restrict__ bvec, float* __restrict__ Hcur,
                        uint16_t* __restrict__ SB, long long total, int relu) {
  long long idx = (long long)blockIdx.x * blockDim.x + threadIdx.x;
  if (idx >= total) return;
  int c = (int)(idx & 127);
  float v = Gout[idx] + bvec[c] + Lb[idx];
  if (relu) v = fmaxf(v, 0.f);
  Hcur[idx] = v;
  if (SB) SB[idx] = f2bf(v);
}

// ---------------------------------------------------------------------------
// z = [h2 | ctrl | pert_emb | 0-pad] -> bf16, Kpad=288.  One wave / node.
// ---------------------------------------------------------------------------
__global__ void build_z(const float* __restrict__ h2, const float* __restrict__ ctrl,
                        const int* __restrict__ pert, const float* __restrict__ pert_table,
                        uint16_t* __restrict__ SB, int N) {
  int wave = (blockIdx.x * blockDim.x + threadIdx.x) >> 5;
  int lane = threadIdx.x & 31;
  if (wave >= N) return;
  const float* hr = h2 + (size_t)wave * 128;
  const float* pr = pert_table + (size_t)pert[wave] * 128;
  float c0 = ctrl[wave];
  uint16_t* row = SB + (size_t)wave * 288;
#pragma unroll
  for (int j = 0; j < 9; ++j) {
    int c = lane + 32 * j;
    float v;
    if (c < 128)      v = hr[c];
    else if (c == 128) v = c0;
    else if (c < 257) v = pr[c - 129];
    else              v = 0.f;
    row[c] = f2bf(v);
  }
}

// ---------------------------------------------------------------------------
// out = relu( relu(M1) @ Wm2 + bm2 ).  One wave / node (64 -> 1).
// ---------------------------------------------------------------------------
__global__ void mlp_out(const float* __restrict__ M1, const float* __restrict__ Wm2,
                        const float* __restrict__ bm2, float* __restrict__ out, int N) {
  int wave = (blockIdx.x * blockDim.x + threadIdx.x) >> 5;
  int lane = threadIdx.x & 31;
  if (wave >= N) return;
  const float* r = M1 + (size_t)wave * 64;
  float s = fmaxf(r[lane], 0.f) * Wm2[lane] + fmaxf(r[lane + 32], 0.f) * Wm2[lane + 32];
#pragma unroll
  for (int off = 16; off; off >>= 1) s += __shfl_down(s, off, 32);
  if (lane == 0) out[wave] = fmaxf(s + bm2[0], 0.f);
}

// ---------------------------------------------------------------------------
extern "C" void kernel_launch(void* const* d_in, const int* in_sizes, int n_in,
                              void* d_out, int out_size, void* d_ws, size_t ws_size,
                              hipStream_t stream) {
  const float* x        = (const float*)d_in[0];
  const int*   eidx     = (const int*)d_in[1];
  // d_in[2] = edge_attr: unused by the reference forward pass
  const int*   pert     = (const int*)d_in[3];
  const float* ctrl     = (const float*)d_in[4];
  const int*   pos      = (const int*)d_in[5];
  const float* gene_tab = (const float*)d_in[6];
  const float* pert_tab = (const float*)d_in[7];
  const float* W1  = (const float*)d_in[8];
  const float* as1 = (const float*)d_in[9];
  const float* ad1 = (const float*)d_in[10];
  const float* b1  = (const float*)d_in[11];
  const float* Wl1 = (const float*)d_in[12];
  const float* bl1 = (const float*)d_in[13];
  const float* W2  = (const float*)d_in[14];
  const float* as2 = (const float*)d_in[15];
  const float* ad2 = (const float*)d_in[16];
  const float* b2  = (const float*)d_in[17];
  const float* Wl2 = (const float*)d_in[18];
  const float* bl2 = (const float*)d_in[19];
  const float* Wm1 = (const float*)d_in[20];
  const float* bm1 = (const float*)d_in[21];
  const float* Wm2 = (const float*)d_in[22];
  const float* bm2 = (const float*)d_in[23];
  float* out = (float*)d_out;

  const int N = in_sizes[0] / 64;   // DF = 64
  const int E = in_sizes[1] / 2;
  const int* srcI = eidx;
  const int* dstI = eidx + E;

  char* wp = (char*)d_ws;
  auto carve = [&](size_t bytes) -> void* {
    void* r = (void*)wp;
    wp += (bytes + 255) & ~(size_t)255;
    return r;
  };
  uint16_t* SB   = (uint16_t*)carve((size_t)N * 288 * 2); // h0(192)/h1(128)/z(288) bf16 staging
  float*    Ha   = (float*)carve((size_t)N * 128 * 4);    // h @ W   (attention branch)
  float*    Lb   = (float*)carve((size_t)N * 128 * 4);    // h @ Wl + bl
  float*    Gout = (float*)carve((size_t)N * 128 * 4);    // attention aggregate
  float*    Hcur = (float*)carve((size_t)N * 128 * 4);    // layer output
  float*    M1   = (float*)carve((size_t)N * 64 * 4);     // MLP hidden
  float*    asrc = (float*)carve((size_t)N * 4);
  float*    adst = (float*)carve((size_t)N * 4);
  unsigned* mkey = (unsigned*)carve((size_t)N * 4);
  float*    deno = (float*)carve((size_t)N * 4);
  uint16_t* W1t  = (uint16_t*)carve((size_t)128 * 192 * 2);
  uint16_t* Wl1t = (uint16_t*)carve((size_t)128 * 192 * 2);
  uint16_t* W2t  = (uint16_t*)carve((size_t)128 * 128 * 2);
  uint16_t* Wl2t = (uint16_t*)carve((size_t)128 * 128 * 2);
  uint16_t* Wm1t = (uint16_t*)carve((size_t)64 * 288 * 2);

  const int B = 256;
  auto cdiv = [](long long a, long long b) { return (int)((a + b - 1) / b); };
  const int nodeWaveGrid = cdiv(N, 8);                  // wave-per-node kernels
  const int stripeGrid   = cdiv(cdiv(N, 16), 8);        // wave-per-16-row-stripe GEMM
  const int edgeGrid     = cdiv((long long)E + N, B);
  int edgeScatGrid = cdiv((long long)E + N, 8);
  if (edgeScatGrid > 12288) edgeScatGrid = 12288;       // grid-stride + prefetch

  // dynamic LDS sizes for the B-matrix stage (row pad +8 halves)
  const size_t lds192 = (size_t)128 * (192 + 8) * 2;    // 51.2 KB
  const size_t lds128 = (size_t)128 * (128 + 8) * 2;    // 34.8 KB
  const size_t lds288 = (size_t)64  * (288 + 8) * 2;    // 37.9 KB

  // --- weight staging (bf16, transposed, K zero-padded to mult of 32) ---
  convert_weight<<<cdiv(128 * 192, B), B, 0, stream>>>(W1,  W1t,  192, 128, 192);
  convert_weight<<<cdiv(128 * 192, B), B, 0, stream>>>(Wl1, Wl1t, 192, 128, 192);
  convert_weight<<<cdiv(128 * 128, B), B, 0, stream>>>(W2,  W2t,  128, 128, 128);
  convert_weight<<<cdiv(128 * 128, B), B, 0, stream>>>(Wl2, Wl2t, 128, 128, 128);
  convert_weight<<<cdiv(64 * 288, B), B, 0, stream>>>(Wm1, Wm1t, 257, 64, 288);

  // --- h0 = [x | renorm(gene_emb)] ---
  build_h0<<<nodeWaveGrid, B, 0, stream>>>(x, pos, gene_tab, SB, N);

  // ================= GAT layer 1 =================
  wmma_gemm_lds<8><<<stripeGrid, B, lds192, stream>>>(SB, W1t,  nullptr, Ha, N, 192);
  wmma_gemm_lds<8><<<stripeGrid, B, lds192, stream>>>(SB, Wl1t, bl1,     Lb, N, 192);
  row_dots<<<nodeWaveGrid, B, 0, stream>>>(Ha, as1, ad1, asrc, adst, N);
  fill_u32<<<cdiv(N, B), B, 0, stream>>>(mkey, 0u, N);
  fill_u32<<<cdiv(N, B), B, 0, stream>>>((unsigned*)deno, 0u, N);
  fill_u32<<<cdiv((long long)N * 128, B), B, 0, stream>>>((unsigned*)Gout, 0u, (long long)N * 128);
  edge_max    <<<edgeGrid,     B, 0, stream>>>(srcI, dstI, asrc, adst, mkey, E, N);
  edge_sum    <<<edgeGrid,     B, 0, stream>>>(srcI, dstI, asrc, adst, mkey, deno, E, N);
  edge_scatter<<<edgeScatGrid, B, 0, stream>>>(srcI, dstI, asrc, adst, mkey, deno, Ha, Gout, E, N);
  // h1 = relu(gat + b1 + Lb); also write bf16 staging (Kpad=128) for layer 2
  combine<<<cdiv((long long)N * 128, B), B, 0, stream>>>(Gout, Lb, b1, Hcur, SB,
                                                         (long long)N * 128, 1);

  // ================= GAT layer 2 =================
  wmma_gemm_lds<8><<<stripeGrid, B, lds128, stream>>>(SB, W2t,  nullptr, Ha, N, 128);
  wmma_gemm_lds<8><<<stripeGrid, B, lds128, stream>>>(SB, Wl2t, bl2,     Lb, N, 128);
  row_dots<<<nodeWaveGrid, B, 0, stream>>>(Ha, as2, ad2, asrc, adst, N);
  fill_u32<<<cdiv(N, B), B, 0, stream>>>(mkey, 0u, N);
  fill_u32<<<cdiv(N, B), B, 0, stream>>>((unsigned*)deno, 0u, N);
  fill_u32<<<cdiv((long long)N * 128, B), B, 0, stream>>>((unsigned*)Gout, 0u, (long long)N * 128);
  edge_max    <<<edgeGrid,     B, 0, stream>>>(srcI, dstI, asrc, adst, mkey, E, N);
  edge_sum    <<<edgeGrid,     B, 0, stream>>>(srcI, dstI, asrc, adst, mkey, deno, E, N);
  edge_scatter<<<edgeScatGrid, B, 0, stream>>>(srcI, dstI, asrc, adst, mkey, deno, Ha, Gout, E, N);
  // h2 = gat + b2 + Lb (no relu before the MLP concat)
  combine<<<cdiv((long long)N * 128, B), B, 0, stream>>>(Gout, Lb, b2, Hcur, nullptr,
                                                         (long long)N * 128, 0);

  // ================= MLP head =================
  build_z<<<nodeWaveGrid, B, 0, stream>>>(Hcur, ctrl, pert, pert_tab, SB, N);
  wmma_gemm_lds<4><<<stripeGrid, B, lds288, stream>>>(SB, Wm1t, bm1, M1, N, 288);
  mlp_out<<<nodeWaveGrid, B, 0, stream>>>(M1, Wm2, bm2, out, N);
}
